// MultiheadAttention_16097537425671
// MI455X (gfx1250) — matrix operand order repaired and checked by reading, not compiled, not run on hardware
//
#include <hip/hip_runtime.h>
#include <hip/hip_bf16.h>

// MI455X (gfx1250) multihead-attention forward.
// f16 WMMA (v_wmma_f32_16x16x32_f16) for all three GEMM stages, f32 softmax,
// attn_weight mean via block-exclusive global RMW (L2-resident), and TDM
// (tensor_load_to_lds, TENSORcnt) double-buffered tile staging into LDS.

typedef _Float16 half_t;
typedef __attribute__((ext_vector_type(16))) _Float16 v16h;
typedef __attribute__((ext_vector_type(8)))  _Float16 v8h;
typedef __attribute__((ext_vector_type(8)))  float    v8f;
typedef __attribute__((ext_vector_type(4)))  unsigned int u32x4;
typedef __attribute__((ext_vector_type(8)))  int      i32x8;
typedef __attribute__((ext_vector_type(4)))  int      i32x4;

union AFrag { v16h v; v8h h8[2]; };

#define WMMA_F16(a, b, c) \
  __builtin_amdgcn_wmma_f32_16x16x32_f16(false, (a), false, (b), (short)0, (c), false, false)

#define SEQ   2048
#define NH    12
#define QT    16
#define LDQKV 2304
#define CDIM  768

// Issue a 2-D f16 tile DMA global->LDS via the Tensor Data Mover.
// D# packing per CDNA5 ISA ch.8: group0 {count, lds_addr, global_addr, type=2},
// group1 {data_size=1(2B), tensor_dim0/1, tile_dim0/1, tensor_dim0_stride}.
// This toolchain exposes the 6-arg builtin: (g0, g1, g2, g3, g_extra, cpol).
__device__ __forceinline__ void tdm_load_2d(unsigned lds_addr, const half_t* gptr,
                                            unsigned tensor_d0, unsigned tensor_d1,
                                            unsigned tile_d0, unsigned tile_d1,
                                            unsigned stride_d0) {
  unsigned long long ga = (unsigned long long)(uintptr_t)gptr;
  u32x4 g0;
  g0.x = 1u;                                              // count=1, user descriptor
  g0.y = lds_addr;                                        // LDS byte address
  g0.z = (unsigned)(ga & 0xffffffffu);                    // global_addr[31:0]
  g0.w = (unsigned)((ga >> 32) & 0x1ffffffu) | (2u << 30); // global_addr[56:32] | type=2
  i32x8 g1;
  g1[0] = 0x00010000;                                     // data_size=1 (2 bytes)
  g1[1] = (int)((tensor_d0 & 0xffffu) << 16);             // tensor_dim0[15:0]
  g1[2] = (int)((tensor_d0 >> 16) | ((tensor_d1 & 0xffffu) << 16));
  g1[3] = (int)((tensor_d1 >> 16) | (tile_d0 << 16));     // tile_dim0
  g1[4] = (int)tile_d1;                                   // tile_dim1 (tile_dim2=0)
  g1[5] = (int)stride_d0;                                 // tensor_dim0_stride[31:0]
  g1[6] = 0;                                              // stride[47:32]=0, dim1_stride lo=0
  g1[7] = 0;
  i32x4 z4 = {0, 0, 0, 0};                                // 2-D tensor: groups 2/3 unused
  i32x8 z8 = {0, 0, 0, 0, 0, 0, 0, 0};
  __builtin_amdgcn_tensor_load_to_lds(g0, g1, z4, z4, z8, 0);
}

__device__ __forceinline__ unsigned lds_off(const void* p) {
  return (unsigned)(uintptr_t)p;  // generic addr low 32 bits == LDS byte offset
}

__global__ __launch_bounds__(256) void f32_to_f16_kernel(const float* __restrict__ in,
                                                         half_t* __restrict__ out, int n) {
  int i = blockIdx.x * 256 + threadIdx.x;
  if (i < n) out[i] = (half_t)in[i];
}

// C[M,N] = A[M,K] * B[N,K]^T ; A,B f16 row-major, C row-major (f16 or f32).
// Block: 256 threads (8 wave32), 64x64 C tile, K step 32, TDM double-buffered staging.
template<int K, typename OutT>
__global__ __launch_bounds__(256) void gemm_nt_kernel(const half_t* __restrict__ A,
                                                      const half_t* __restrict__ B,
                                                      OutT* __restrict__ C, int N) {
  __shared__ half_t As[2][64 * 32];
  __shared__ half_t Bs[2][64 * 32];
  const int tid  = threadIdx.x;
  const int lane = tid & 31, wave = tid >> 5;
  const int nTilesN = N >> 6;
  const int mt = blockIdx.x / nTilesN, nt = blockIdx.x % nTilesN;
  const int mw = wave & 3;      // 4 M-subtiles of 16
  const int nw = wave >> 2;     // 2 N-halves of 32 (each wave: two 16x16 N-subtiles)
  const int c0a = (lane < 16) ? 0 : 16;  // A-frag K split (lanes 0-15: K 0-15)
  const int bko = (lane < 16) ? 0 : 16;  // B-frag K split (lanes 0-15: K 0-15)
  const int l15 = lane & 15;
  const int mloc = mw * 16 + l15;
  const int nloc = nw * 32 + l15;
  const half_t* Abase = A + (size_t)(mt * 64) * K;
  const half_t* Bbase = B + (size_t)(nt * 64) * K;
  v8f acc0 = {}, acc1 = {};
  if (wave == 0) {  // prologue: DMA tile 0
    tdm_load_2d(lds_off(&As[0][0]), Abase, K, 64, 32, 64, K);
    tdm_load_2d(lds_off(&Bs[0][0]), Bbase, K, 64, 32, 64, K);
  }
  for (int k0 = 0, cur = 0; k0 < K; k0 += 32, cur ^= 1) {
    if (wave == 0) __builtin_amdgcn_s_wait_tensorcnt(0);
    __syncthreads();  // tile `cur` visible to all waves; prior reads of `cur^1` done
    if (wave == 0 && k0 + 32 < K) {  // prefetch next tile into the other buffer
      tdm_load_2d(lds_off(&As[cur ^ 1][0]), Abase + k0 + 32, K, 64, 32, 64, K);
      tdm_load_2d(lds_off(&Bs[cur ^ 1][0]), Bbase + k0 + 32, K, 64, 32, 64, K);
    }
    const half_t* curA = As[cur];
    const half_t* curB = Bs[cur];
    AFrag a;
    a.h8[0] = *(const v8h*)&curA[mloc * 32 + c0a];
    a.h8[1] = *(const v8h*)&curA[mloc * 32 + c0a + 8];
    v16h b0 = *(const v16h*)&curB[nloc * 32 + bko];
    v16h b1 = *(const v16h*)&curB[(nloc + 16) * 32 + bko];
    acc0 = WMMA_F16(a.v, b0, acc0);
    acc1 = WMMA_F16(a.v, b1, acc1);
  }
  // C/D layout: element (M = r + 8*(lane>=16), N = lane&15) in vgpr r
  const int mbase = mt * 64 + mw * 16 + 8 * (lane >> 4);
  const int nbase = nt * 64 + nw * 32 + l15;
  #pragma unroll
  for (int r = 0; r < 8; ++r) {
    C[(size_t)(mbase + r) * N + nbase]      = (OutT)acc0[r];
    C[(size_t)(mbase + r) * N + nbase + 16] = (OutT)acc1[r];
  }
}

// One block per (batch, 16-row q-tile); loops over all 12 heads.
// LDS: sc[16][2048] f32 scores (softmaxed in place, f16 probs within each row's
// own region), vtile[2][64][64] f16 V double buffer (TDM), cred[4][256] reduce.
__global__ __launch_bounds__(256) void attn_kernel(const half_t* __restrict__ qkv,
                                                   float* __restrict__ aw,
                                                   half_t* __restrict__ attnh) {
  extern __shared__ float smem_f[];
  float*  sc    = smem_f;                                   // 131072 B
  half_t* vtile = (half_t*)((char*)smem_f + QT * SEQ * 4);  // 2 x 8192 B
  float*  cred  = (float*)((char*)vtile + 2 * 64 * 64 * 2); // 4096 B

  const int tid  = threadIdx.x;
  const int lane = tid & 31, wave = tid >> 5;
  const int b  = blockIdx.x / (SEQ / QT);
  const int qt = blockIdx.x % (SEQ / QT);
  const int c0a = (lane < 16) ? 0 : 16;
  const int bko = (lane < 16) ? 0 : 16;
  const int l15 = lane & 15;
  const int hi  = lane >> 4;

  for (int h = 0; h < NH; ++h) {
    const int qc0 = h * 192;  // q at +0, k at +64, v at +128 within a head's columns
    const half_t* Vbase = qkv + (size_t)(b * SEQ) * LDQKV + qc0 + 128;
    // ---- Q fragments for this head: A-matrix 16x64 as two 16x32 chunks ----
    AFrag qa0, qa1;
    {
      const size_t qrow = (size_t)(b * SEQ + qt * QT + l15) * LDQKV;
      qa0.h8[0] = *(const v8h*)&qkv[qrow + qc0 + c0a];
      qa0.h8[1] = *(const v8h*)&qkv[qrow + qc0 + c0a + 8];
      qa1.h8[0] = *(const v8h*)&qkv[qrow + qc0 + 32 + c0a];
      qa1.h8[1] = *(const v8h*)&qkv[qrow + qc0 + 32 + c0a + 8];
    }
    // ---- phase 1: scores = (Q K^T) * 1/sqrt(64), 16-key tiles striped over waves ----
    for (int kt = wave; kt < SEQ / 16; kt += 8) {
      const size_t krow = (size_t)(b * SEQ + kt * 16 + l15) * LDQKV;
      v16h bf0 = *(const v16h*)&qkv[krow + qc0 + 64 + bko];
      v16h bf1 = *(const v16h*)&qkv[krow + qc0 + 64 + 32 + bko];
      v8f s = {};
      s = WMMA_F16(qa0.v, bf0, s);
      s = WMMA_F16(qa1.v, bf1, s);
      #pragma unroll
      for (int r = 0; r < 8; ++r)
        sc[(r + 8 * hi) * SEQ + kt * 16 + l15] = s[r] * 0.125f;
    }
    __syncthreads();
    // Overlap: DMA first V tile of this head while softmax runs (vtile idle since
    // the barrier that ended the previous head's phase 3).
    if (wave == 0) tdm_load_2d(lds_off(&vtile[0]), Vbase, LDQKV, SEQ, 64, 64, LDQKV);
    // ---- phase 2: row softmax + attn_weight accumulation + f16 probs in place ----
    #pragma unroll
    for (int rr = 0; rr < 2; ++rr) {
      const int r = wave * 2 + rr;
      float* row = sc + r * SEQ;
      float m = -3.0e38f;
      for (int t = lane; t < SEQ; t += 32) m = fmaxf(m, row[t]);
      #pragma unroll
      for (int off = 16; off > 0; off >>= 1) m = fmaxf(m, __shfl_xor(m, off, 32));
      float ssum = 0.f;
      for (int t = lane; t < SEQ; t += 32) {
        float e = __expf(row[t] - m);
        row[t] = e;
        ssum += e;
      }
      #pragma unroll
      for (int off = 16; off > 0; off >>= 1) ssum += __shfl_xor(ssum, off, 32);
      const float inv = 1.f / ssum;
      half_t* ph = (half_t*)row;  // f16 probs occupy first half of this row's region
      float* awrow = aw + ((size_t)b * SEQ + qt * QT + r) * SEQ;
      for (int t = lane; t < SEQ; t += 32) {
        float p = row[t] * inv;
        ph[t] = (half_t)p;                       // write idx t/2 <= read idx t: safe in-order
        float prev = (h == 0) ? 0.f : awrow[t];  // block-exclusive rows; h==0 resets
        awrow[t] = prev + p * (1.f / 12.f);
      }
    }
    __syncthreads();
    // ---- phase 3: attn_out(16x64) = P(16x2048) @ V(2048x64), TDM-pipelined V ----
    const int nsub = wave & 3;   // which 16-wide d-subtile
    const int kh   = wave >> 2;  // which 32-key half of each staged 64-key tile
    const half_t* ph = (const half_t*)sc;  // probs row r at ph[r*4096 + key]
    const int d = nsub * 16 + l15;
    v8f o = {};
    for (int kt4 = 0, cur = 0; kt4 < SEQ / 64; ++kt4, cur ^= 1) {
      if (wave == 0) __builtin_amdgcn_s_wait_tensorcnt(0);
      __syncthreads();  // V tile `cur` ready; prior reads of `cur^1` complete
      if (wave == 0 && kt4 + 1 < SEQ / 64)
        tdm_load_2d(lds_off(&vtile[(cur ^ 1) * 64 * 64]), Vbase + (size_t)(kt4 + 1) * 64 * LDQKV,
                    LDQKV, SEQ, 64, 64, LDQKV);
      const half_t* vt = vtile + cur * 64 * 64;
      const int kb = kt4 * 64 + kh * 32;
      AFrag pa;
      pa.h8[0] = *(const v8h*)&ph[l15 * 4096 + kb + c0a];
      pa.h8[1] = *(const v8h*)&ph[l15 * 4096 + kb + c0a + 8];
      v16h bv;
      const int kl0 = kh * 32 + bko;
      #pragma unroll
      for (int j = 0; j < 16; ++j) bv[j] = vt[(kl0 + j) * 64 + d];
      o = WMMA_F16(pa.v, bv, o);
    }
    // reduce the two K-halves (waves kh==1 -> LDS, waves kh==0 add + store)
    if (kh == 1) {
      #pragma unroll
      for (int r = 0; r < 8; ++r) cred[nsub * 256 + r * 32 + lane] = o[r];
    }
    __syncthreads();
    if (kh == 0) {
      #pragma unroll
      for (int r = 0; r < 8; ++r) {
        float v = o[r] + cred[nsub * 256 + r * 32 + lane];
        attnh[(size_t)(b * SEQ + qt * QT + r + 8 * hi) * CDIM + h * 64 + nsub * 16 + l15] =
            (half_t)v;
      }
    }
    __syncthreads();  // before next head reuses sc/cred/vtile
  }
}

extern "C" void kernel_launch(void* const* d_in, const int* in_sizes, int n_in,
                              void* d_out, int out_size, void* d_ws, size_t ws_size,
                              hipStream_t stream) {
  (void)in_sizes; (void)n_in; (void)out_size; (void)ws_size;
  const float* x     = (const float*)d_in[0];  // [2,2048,768]
  const float* w_qkv = (const float*)d_in[1];  // [2304,768]
  const float* w_out = (const float*)d_in[2];  // [768,768]
  float* out = (float*)d_out;                  // [4096,768]
  float* aw  = out + (size_t)4096 * CDIM;      // [2,2048,2048]

  char* ws = (char*)d_ws;
  size_t off = 0;
  auto carve = [&](size_t bytes) -> char* {
    char* p = ws + off;
    off += (bytes + 255) & ~(size_t)255;
    return p;
  };
  half_t* xh   = (half_t*)carve((size_t)4096 * CDIM * 2);
  half_t* wqh  = (half_t*)carve((size_t)LDQKV * CDIM * 2);
  half_t* woh  = (half_t*)carve((size_t)CDIM * CDIM * 2);
  half_t* qkvh = (half_t*)carve((size_t)4096 * LDQKV * 2);
  half_t* atth = (half_t*)carve((size_t)4096 * CDIM * 2);

  const int nx = 4096 * CDIM, nwq = LDQKV * CDIM, nwo = CDIM * CDIM;
  f32_to_f16_kernel<<<(nx + 255) / 256, 256, 0, stream>>>(x, xh, nx);
  f32_to_f16_kernel<<<(nwq + 255) / 256, 256, 0, stream>>>(w_qkv, wqh, nwq);
  f32_to_f16_kernel<<<(nwo + 255) / 256, 256, 0, stream>>>(w_out, woh, nwo);

  // qkv = x @ w_qkv^T : M=4096, N=2304, K=768 -> f16
  gemm_nt_kernel<CDIM, half_t><<<(4096 / 64) * (LDQKV / 64), 256, 0, stream>>>(xh, wqh, qkvh, LDQKV);

  // fused attention: 2 batches * 128 q-tiles
  const size_t attn_smem = (size_t)QT * SEQ * 4 + 2 * 64 * 64 * 2 + 4 * 256 * 4;  // 151552 B
  attn_kernel<<<2 * (SEQ / QT), 256, attn_smem, stream>>>(qkvh, aw, atth);

  // out = attn @ w_out^T : M=4096, N=768, K=768 -> f32
  gemm_nt_kernel<CDIM, float><<<(4096 / 64) * (CDIM / 64), 256, 0, stream>>>(atth, woh, out, CDIM);
}
